// EncoderWithLSTM_49031346651728
// MI455X (gfx1250) — compile-verified, matrix operand order checked by reading, not run on hardware
//
#include <hip/hip_runtime.h>

// ---------------------------------------------------------------------------
// 2-layer LSTM single step, M = P*B = 131072 rows, H = 128, gates = 512.
// Pre-pass converts weights to packed bf16 hi/lo (fragment-ready) in d_ws.
// Main kernel: bf16-split 3-product WMMA f32_16x16x32_bf16, f32 accumulate.
// Weight slabs staged to LDS with gfx1250 async global->LDS loads.
// ---------------------------------------------------------------------------

#define P_    64
#define B_    2048
#define H_    128
#define IN_   2
#define GATES 512                 // 4*H
#define ASTR  132                 // f32 LDS stride for h1 staging (bank-friendly)
#define BSTR  68                  // packed-dword LDS stride for weight slabs
#define WDW   (GATES * (H_ / 2))  // 32768 dwords per packed half-matrix

typedef __attribute__((ext_vector_type(16))) __bf16 bf16x16;
typedef __attribute__((ext_vector_type(8)))  float  f32x8;
typedef int v4i_ __attribute__((vector_size(16)));

#if __has_builtin(__builtin_amdgcn_global_load_async_to_lds_b128)
#define HAS_ASYNC_LDS 1
#else
#define HAS_ASYNC_LDS 0
#endif

// ---- bf16 split helpers ----------------------------------------------------
__device__ __forceinline__ unsigned bf16_rne_bits(float x) {
  unsigned u = __builtin_bit_cast(unsigned, x);
  return (u + 0x7FFFu + ((u >> 16) & 1u)) >> 16;
}

// two f32 -> packed hi dword {bf16(a),bf16(b)} and packed lo dword (residuals)
__device__ __forceinline__ void split_pack_pair(float a, float b,
                                                unsigned& hiw, unsigned& low) {
  unsigned ha = bf16_rne_bits(a), hb = bf16_rne_bits(b);
  float ra = __builtin_bit_cast(float, ha << 16);
  float rb = __builtin_bit_cast(float, hb << 16);
  unsigned la = bf16_rne_bits(a - ra), lb = bf16_rne_bits(b - rb);
  hiw = ha | (hb << 16);
  low = la | (lb << 16);
}

// 16 f32 fragment elements -> hi/lo bf16 fragments (A-side only, rare)
__device__ __forceinline__ void split16(const float* v, bf16x16& hi, bf16x16& lo) {
  union U { bf16x16 v; unsigned u[8]; } H, L;
#pragma unroll
  for (int i = 0; i < 8; ++i) split_pack_pair(v[2 * i], v[2 * i + 1], H.u[i], L.u[i]);
  hi = H.v;
  lo = L.v;
}

// A-side: load 16 f32 fragment elements for one lane
// e=0..7 -> rowBase[k0..k0+7], e=8..15 -> rowBase[k0+16..k0+23]
__device__ __forceinline__ void load16f(const float* rowBase, int k0, float* v) {
  float4 a = *reinterpret_cast<const float4*>(rowBase + k0 + 0);
  float4 b = *reinterpret_cast<const float4*>(rowBase + k0 + 4);
  float4 c = *reinterpret_cast<const float4*>(rowBase + k0 + 16);
  float4 d = *reinterpret_cast<const float4*>(rowBase + k0 + 20);
  v[0] = a.x;  v[1] = a.y;  v[2] = a.z;  v[3] = a.w;
  v[4] = b.x;  v[5] = b.y;  v[6] = b.z;  v[7] = b.w;
  v[8] = c.x;  v[9] = c.y;  v[10] = c.z; v[11] = c.w;
  v[12] = d.x; v[13] = d.y; v[14] = d.z; v[15] = d.w;
}

// B-side: packed bf16 fragment straight from LDS (two b128, zero VALU)
__device__ __forceinline__ bf16x16 loadBfrag(const unsigned* row, int d0) {
  union { bf16x16 v; uint4 q[2]; } F;
  F.q[0] = *reinterpret_cast<const uint4*>(row + d0);
  F.q[1] = *reinterpret_cast<const uint4*>(row + d0 + 8);
  return F.v;
}

// stage one 16-byte granule global -> LDS (async on gfx1250 toolchains)
__device__ __forceinline__ void stage16B(const unsigned* gsrc, unsigned* ldst) {
#if HAS_ASYNC_LDS
  __builtin_amdgcn_global_load_async_to_lds_b128(
      (__attribute__((address_space(1))) v4i_*)gsrc,
      (__attribute__((address_space(3))) v4i_*)ldst, 0, 0);
#else
  *reinterpret_cast<uint4*>(ldst) = *reinterpret_cast<const uint4*>(gsrc);
#endif
}

__device__ __forceinline__ void stage_fence() {
#if HAS_ASYNC_LDS
#if __has_builtin(__builtin_amdgcn_s_wait_asynccnt)
  __builtin_amdgcn_s_wait_asynccnt(0);
#else
  asm volatile("s_wait_asynccnt 0x0" ::: "memory");
#endif
#endif
}

// 3-product split MMA: acc += (Ah+Al)*(Bh+Bl) (dropping Al*Bl ~ 2^-16)
__device__ __forceinline__ void mma3(const bf16x16& ah, const bf16x16& al,
                                     const bf16x16& bh, const bf16x16& bl,
                                     f32x8& acc) {
  acc = __builtin_amdgcn_wmma_f32_16x16x32_bf16(false, ah, false, bh, (short)0, acc, false, false);
  acc = __builtin_amdgcn_wmma_f32_16x16x32_bf16(false, ah, false, bl, (short)0, acc, false, false);
  acc = __builtin_amdgcn_wmma_f32_16x16x32_bf16(false, al, false, bh, (short)0, acc, false, false);
}

// ---- transcendentals (gfx1250 V_TANH_F32) ---------------------------------
__device__ __forceinline__ float fast_tanh(float x) {
#if __has_builtin(__builtin_amdgcn_tanhf)
  return __builtin_amdgcn_tanhf(x);
#elif __has_builtin(__builtin_amdgcn_tanh_f32)
  return __builtin_amdgcn_tanh_f32(x);
#else
  float y;
  asm("v_tanh_f32 %0, %1\n\tv_nop\n\tv_nop" : "=v"(y) : "v"(x));
  return y;
#endif
}
__device__ __forceinline__ float sigm(float x) {
  return fast_tanh(0.5f * x) * 0.5f + 0.5f;
}

// ---------------------------------------------------------------------------
// Pre-pass: W (512x128 f32, row-major) -> packed hi[32768 dw] + lo[32768 dw]
// ws layout: [m=0 Whh0][m=1 Wih1][m=2 Whh1], each {hi: WDW dw, lo: WDW dw}
// ---------------------------------------------------------------------------
__global__ __launch_bounds__(256) void prep_weights_kernel(
    const float* __restrict__ Whh0, const float* __restrict__ Wih1,
    const float* __restrict__ Whh1, unsigned* __restrict__ ws) {
  int idx = blockIdx.x * 256 + threadIdx.x;
  if (idx >= 3 * WDW) return;
  int m = idx / WDW;
  int r = idx - m * WDW;
  const float* W = (m == 0) ? Whh0 : (m == 1) ? Wih1 : Whh1;
  float a = W[2 * r + 0], b = W[2 * r + 1];
  unsigned h, l;
  split_pack_pair(a, b, h, l);
  ws[(size_t)m * 2 * WDW + r]       = h;
  ws[(size_t)m * 2 * WDW + WDW + r] = l;
}

// ---------------------------------------------------------------------------
__global__ __launch_bounds__(256) void lstm2_wmma_kernel(
    const float* __restrict__ x,     // (B, P, IN)
    const float* __restrict__ h0,    // (P, 2, B, H)
    const float* __restrict__ c0,    // (P, 2, B, H)
    const float* __restrict__ Wih0,  // (512, 2)
    const float* __restrict__ bih0, const float* __restrict__ bhh0,
    const float* __restrict__ bih1, const float* __restrict__ bhh1,
    const unsigned* __restrict__ ws, // packed weights (see prep)
    float* __restrict__ enc,         // (B, P, H)
    float* __restrict__ hN,          // (P, 2, B, H)
    float* __restrict__ cN)          // (P, 2, B, H)
{
  extern __shared__ unsigned smemu[];
  unsigned* ldsBHi = smemu;                       // 128 rows * BSTR dwords
  unsigned* ldsBLo = ldsBHi + 128 * BSTR;         // 128 rows * BSTR dwords
  float* ldsH1   = reinterpret_cast<float*>(ldsBLo + 128 * BSTR);  // 8*16*ASTR f32
  float* ldsW0   = ldsH1 + 8 * 16 * ASTR;         // 512*2 f32 (W_ih0)
  float* ldsBias = ldsW0 + GATES * IN_;           // 2*512 f32 (bias sums)

  const unsigned* whh0Hi = ws;
  const unsigned* whh0Lo = ws + WDW;
  const unsigned* wih1Hi = ws + 2 * WDW;
  const unsigned* wih1Lo = ws + 3 * WDW;
  const unsigned* whh1Hi = ws + 4 * WDW;
  const unsigned* whh1Lo = ws + 5 * WDW;

  const int tid  = threadIdx.x;
  const int wave = tid >> 5;
  const int lane = tid & 31;
  const int nl   = lane & 15;  // fragment row (A) / column (B,C,D)
  const int lh   = lane >> 4;  // lane half

  // one-time block staging: bias sums + W_ih0
  for (int i = tid; i < GATES; i += 256) {
    ldsBias[i]         = bih0[i] + bhh0[i];
    ldsBias[GATES + i] = bih1[i] + bhh1[i];
  }
  for (int i = tid; i < GATES * IN_; i += 256) ldsW0[i] = Wih0[i];

  const int stripBase = blockIdx.x * 128 + wave * 16;  // global row (= p*B + b)
  const int p  = stripBase >> 11;                      // / B
  const int b0 = stripBase & (B_ - 1);

  // x rows for this strip (half-wave-uniform addresses per r -> L1 broadcast)
  float xv0[8], xv1[8];
#pragma unroll
  for (int r = 0; r < 8; ++r) {
    const float* xp = x + ((size_t)(b0 + r + lh * 8) * P_ + p) * IN_;
    xv0[r] = xp[0];
    xv1[r] = xp[1];
  }

  // =========================== layer 0 ===========================
  bf16x16 AhHi[4], AhLo[4];  // hidden-state A fragments (reused for layer 1)
  {
    const float* aRow = h0 + ((size_t)(p * 2 + 0) * B_ + b0 + nl) * H_;
#pragma unroll
    for (int kc = 0; kc < 4; ++kc) {
      float v[16];
      load16f(aRow, kc * 32 + lh * 8, v);
      split16(v, AhHi[kc], AhLo[kc]);
    }
  }

  const float* cin0 = c0 + ((size_t)(p * 2 + 0) * B_ + b0 + lh * 8) * H_;
  float* hOut0 = hN + ((size_t)(p * 2 + 0) * B_ + b0 + lh * 8) * H_;
  float* cOut0 = cN + ((size_t)(p * 2 + 0) * B_ + b0 + lh * 8) * H_;

  for (int nc = 0; nc < 8; ++nc) {
    __syncthreads();
    // stage packed W_hh0 hi/lo slabs: 64 rows (4 gates x 16 cols)
    for (int idx = tid; idx < 64 * 16; idx += 256) {
      int rl = idx >> 4, q4 = (idx & 15) * 4;
      int rowg = (((rl >> 4) * H_ + nc * 16 + (rl & 15)) << 6) + q4;  // *64 dw
      stage16B(whh0Hi + rowg, ldsBHi + rl * BSTR + q4);
      stage16B(whh0Lo + rowg, ldsBLo + rl * BSTR + q4);
    }
    stage_fence();
    __syncthreads();

    f32x8 acc[4];
#pragma unroll
    for (int g = 0; g < 4; ++g) {
      const int n = g * H_ + nc * 16 + nl;
      const float bias = ldsBias[n];
      const float w0 = ldsW0[n * 2 + 0], w1 = ldsW0[n * 2 + 1];
#pragma unroll
      for (int r = 0; r < 8; ++r)  // bias + x@W_ih0^T (K=2, exact f32)
        acc[g][r] = bias + xv0[r] * w0 + xv1[r] * w1;
#pragma unroll
      for (int kc = 0; kc < 4; ++kc) {
        const int d0 = kc * 16 + lh * 4;
        bf16x16 Bhi = loadBfrag(ldsBHi + (g * 16 + nl) * BSTR, d0);
        bf16x16 Blo = loadBfrag(ldsBLo + (g * 16 + nl) * BSTR, d0);
        mma3(AhHi[kc], AhLo[kc], Bhi, Blo, acc[g]);
      }
    }

    // elementwise LSTM for these 16 hidden units
    const int n = nc * 16 + nl;
#pragma unroll
    for (int r = 0; r < 8; ++r) {
      float cp = cin0[(size_t)r * H_ + n];
      float iv = sigm(acc[0][r]);
      float fv = sigm(acc[1][r]);
      float gv = fast_tanh(acc[2][r]);
      float ov = sigm(acc[3][r]);
      float cn = fv * cp + iv * gv;
      float hn = ov * fast_tanh(cn);
      cOut0[(size_t)r * H_ + n] = cn;
      hOut0[(size_t)r * H_ + n] = hn;
      ldsH1[(wave * 16 + r + lh * 8) * ASTR + n] = hn;  // layer-1 input
    }
  }

  // =========================== layer 1 ===========================
  bf16x16 AiHi[4], AiLo[4];  // layer-0 output (same-wave LDS ordering)
  {
    const float* aRow = ldsH1 + (wave * 16 + nl) * ASTR;
#pragma unroll
    for (int kc = 0; kc < 4; ++kc) {
      float v[16];
      load16f(aRow, kc * 32 + lh * 8, v);
      split16(v, AiHi[kc], AiLo[kc]);
    }
    const float* hRow = h0 + ((size_t)(p * 2 + 1) * B_ + b0 + nl) * H_;
#pragma unroll
    for (int kc = 0; kc < 4; ++kc) {
      float v[16];
      load16f(hRow, kc * 32 + lh * 8, v);
      split16(v, AhHi[kc], AhLo[kc]);
    }
  }

  const float* cin1 = c0 + ((size_t)(p * 2 + 1) * B_ + b0 + lh * 8) * H_;
  float* hOut1 = hN + ((size_t)(p * 2 + 1) * B_ + b0 + lh * 8) * H_;
  float* cOut1 = cN + ((size_t)(p * 2 + 1) * B_ + b0 + lh * 8) * H_;
  float* encOut = enc + ((size_t)(b0 + lh * 8) * P_ + p) * H_;

  for (int nc = 0; nc < 8; ++nc) {
    __syncthreads();
    // stage packed W_ih1 (rows 0..63) + W_hh1 (rows 64..127) hi/lo slabs
    for (int idx = tid; idx < 128 * 16; idx += 256) {
      int rl = idx >> 4, q4 = (idx & 15) * 4;
      int r2 = rl & 63;
      const unsigned* baseH = (rl < 64) ? wih1Hi : whh1Hi;
      const unsigned* baseL = (rl < 64) ? wih1Lo : whh1Lo;
      int rowg = (((r2 >> 4) * H_ + nc * 16 + (r2 & 15)) << 6) + q4;
      stage16B(baseH + rowg, ldsBHi + rl * BSTR + q4);
      stage16B(baseL + rowg, ldsBLo + rl * BSTR + q4);
    }
    stage_fence();
    __syncthreads();

    f32x8 acc[4];
#pragma unroll
    for (int g = 0; g < 4; ++g) {
      const int n = g * H_ + nc * 16 + nl;
      const float bias = ldsBias[GATES + n];
#pragma unroll
      for (int r = 0; r < 8; ++r) acc[g][r] = bias;
#pragma unroll
      for (int kc = 0; kc < 4; ++kc) {  // h1 @ W_ih1^T
        const int d0 = kc * 16 + lh * 4;
        bf16x16 Bhi = loadBfrag(ldsBHi + (g * 16 + nl) * BSTR, d0);
        bf16x16 Blo = loadBfrag(ldsBLo + (g * 16 + nl) * BSTR, d0);
        mma3(AiHi[kc], AiLo[kc], Bhi, Blo, acc[g]);
      }
#pragma unroll
      for (int kc = 0; kc < 4; ++kc) {  // h0[:,1] @ W_hh1^T
        const int d0 = kc * 16 + lh * 4;
        bf16x16 Bhi = loadBfrag(ldsBHi + (64 + g * 16 + nl) * BSTR, d0);
        bf16x16 Blo = loadBfrag(ldsBLo + (64 + g * 16 + nl) * BSTR, d0);
        mma3(AhHi[kc], AhLo[kc], Bhi, Blo, acc[g]);
      }
    }

    const int n = nc * 16 + nl;
#pragma unroll
    for (int r = 0; r < 8; ++r) {
      float cp = cin1[(size_t)r * H_ + n];
      float iv = sigm(acc[0][r]);
      float fv = sigm(acc[1][r]);
      float gv = fast_tanh(acc[2][r]);
      float ov = sigm(acc[3][r]);
      float cn = fv * cp + iv * gv;
      float hn = ov * fast_tanh(cn);
      cOut1[(size_t)r * H_ + n] = cn;
      hOut1[(size_t)r * H_ + n] = hn;
      encOut[(size_t)r * P_ * H_ + n] = hn;  // encoder_traces == h2
    }
  }
}

// ---------------------------------------------------------------------------
extern "C" void kernel_launch(void* const* d_in, const int* in_sizes, int n_in,
                              void* d_out, int out_size, void* d_ws, size_t ws_size,
                              hipStream_t stream) {
  const float* x    = (const float*)d_in[0];
  const float* h0   = (const float*)d_in[1];
  const float* c0   = (const float*)d_in[2];
  const float* Wih0 = (const float*)d_in[3];
  const float* Whh0 = (const float*)d_in[4];
  const float* bih0 = (const float*)d_in[5];
  const float* bhh0 = (const float*)d_in[6];
  const float* Wih1 = (const float*)d_in[7];
  const float* Whh1 = (const float*)d_in[8];
  const float* bih1 = (const float*)d_in[9];
  const float* bhh1 = (const float*)d_in[10];

  float* out = (float*)d_out;
  float* enc = out;                            // (B, P, H)
  float* hN  = enc + (size_t)B_ * P_ * H_;     // (P, 2, B, H)
  float* cN  = hN + (size_t)P_ * 2 * B_ * H_;  // (P, 2, B, H)
  unsigned* ws = (unsigned*)d_ws;              // 6 * WDW dwords = 768 KB

  // pre-pass: pack weights to bf16 hi/lo fragment format
  {
    dim3 grid((3 * WDW + 255) / 256), block(256);
    hipLaunchKernelGGL(prep_weights_kernel, grid, block, 0, stream,
                       Whh0, Wih1, Whh1, ws);
  }

  const size_t smemBytes =
      (size_t)(2 * 128 * BSTR + 8 * 16 * ASTR + GATES * IN_ + 2 * GATES) * 4u;
  dim3 grid((P_ * B_) / 128);  // 1024 blocks, 128 rows each
  dim3 block(256);             // 8 wave32 waves
  hipLaunchKernelGGL(lstm2_wmma_kernel, grid, block, smemBytes, stream,
                     x, h0, c0, Wih0, bih0, bhh0, bih1, bhh1, ws,
                     enc, hN, cN);
}
